// SlidingWindowAttention_50603304681856
// MI455X (gfx1250) — compile-verified, matrix operand order checked by reading, not run on hardware
//
#include <hip/hip_runtime.h>

// Sliding-window attention (B=4,H=16,L=4096,D=128,w=512), fp32 in/out.
// Flash-style online softmax, BF16 WMMA on gfx1250 (wave32).
// S computed transposed (S^T = K*Q^T) so P^T is already in B-fragment layout
// for O^T = V^T * P^T -- zero cross-lane data movement between the two GEMMs.

#define L_   4096
#define D_   128
#define W_   512
#define BQ   128     // query rows per block (8 waves * 16)
#define WQ   16      // query rows per wave
#define SKS  136     // LDS row stride (bf16 elems) for K  : 128 + 8 pad
#define SVS  520     // LDS row stride (bf16 elems) for V^T: 512 + 8 pad

typedef __bf16 v16bf __attribute__((ext_vector_type(16)));
typedef float  v8f   __attribute__((ext_vector_type(8)));

union Frag {
  v16bf v;
  uint4 q[2];
  unsigned short s[16];
};

__device__ __forceinline__ unsigned short f2bf_u(float f) {
  unsigned int u = __float_as_uint(f);
  u += 0x7FFFu + ((u >> 16) & 1u);   // round-to-nearest-even
  return (unsigned short)(u >> 16);
}
__device__ __forceinline__ unsigned int pack2(float a, float b) {
  return (unsigned int)f2bf_u(a) | ((unsigned int)f2bf_u(b) << 16);
}

__global__ __launch_bounds__(256) void swa_wmma_kernel(
    const float* __restrict__ Q, const float* __restrict__ K,
    const float* __restrict__ V, float* __restrict__ O) {
  extern __shared__ unsigned short smem[];
  unsigned short* sK  = smem;             // [512][SKS]  K rows, bf16
  unsigned short* sVT = smem + W_ * SKS;  // [128][SVS]  V transposed, bf16

  const int tid  = threadIdx.x;
  const int bh   = blockIdx.y;
  const size_t head = (size_t)bh * L_ * D_;
  const float* kg = K + head + (size_t)(L_ - W_) * D_;
  const float* vg = V + head + (size_t)(L_ - W_) * D_;

  // ---- Phase 1: stage K (row-major) and V^T into LDS as bf16 ----
  // 512 rows * 32 float4 = 16384 vec4 loads, 64 per thread.
  for (int it = 0; it < 64; ++it) {
    const int i   = tid + it * 256;
    const int row = i >> 5;       // key row 0..511
    const int c4  = i & 31;       // float4 column
    const float4 kv = *(const float4*)(kg + row * D_ + c4 * 4);
    unsigned int* dk = (unsigned int*)(sK + row * SKS + c4 * 4);
    dk[0] = pack2(kv.x, kv.y);
    dk[1] = pack2(kv.z, kv.w);
    const float4 vv = *(const float4*)(vg + row * D_ + c4 * 4);
    const int d = c4 * 4;
    sVT[(d + 0) * SVS + row] = f2bf_u(vv.x);
    sVT[(d + 1) * SVS + row] = f2bf_u(vv.y);
    sVT[(d + 2) * SVS + row] = f2bf_u(vv.z);
    sVT[(d + 3) * SVS + row] = f2bf_u(vv.w);
  }
  __syncthreads();

  const int wave = tid >> 5;
  const int lane = tid & 31;
  const int h    = lane >> 4;   // lane half (K-split dim of fragments)
  const int idx  = lane & 15;   // row-within-fragment / query column
  const int q0   = blockIdx.x * BQ + wave * WQ;
  const int qrow = q0 + idx;

  // ---- Phase 2: Q fragments in B-matrix layout (B[d][m] = Q[m][d]) ----
  // lane(h,idx) holds query row `qrow`; elems 0..7 = d in {db+8h..+7},
  // elems 8..15 = d in {db+16+8h..+7}.
  v16bf qa[4];
  {
    const float* gq = Q + head + (size_t)qrow * D_;
#pragma unroll
    for (int c = 0; c < 4; ++c) {
      const int db = 32 * c + 8 * h;
      const float4 x = *(const float4*)(gq + db);
      const float4 y = *(const float4*)(gq + db + 4);
      const float4 z = *(const float4*)(gq + db + 16);
      const float4 w = *(const float4*)(gq + db + 20);
      Frag f;
      f.s[0]  = f2bf_u(x.x); f.s[1]  = f2bf_u(x.y);
      f.s[2]  = f2bf_u(x.z); f.s[3]  = f2bf_u(x.w);
      f.s[4]  = f2bf_u(y.x); f.s[5]  = f2bf_u(y.y);
      f.s[6]  = f2bf_u(y.z); f.s[7]  = f2bf_u(y.w);
      f.s[8]  = f2bf_u(z.x); f.s[9]  = f2bf_u(z.y);
      f.s[10] = f2bf_u(z.z); f.s[11] = f2bf_u(z.w);
      f.s[12] = f2bf_u(w.x); f.s[13] = f2bf_u(w.y);
      f.s[14] = f2bf_u(w.z); f.s[15] = f2bf_u(w.w);
      qa[c] = f.v;
    }
  }

  float m_i = -1e30f, l_i = 0.0f;
  v8f o[8] = {};   // O^T accumulators: tile t -> d in [16t,16t+16), query = lane&15
  const float scale = 0.08838834764831845f;  // 1/sqrt(128)
  // Chunks with n0 > q0+15 are fully masked for this wave -> skip.
  int n_end = (((q0 + WQ - 1) >> 5) + 1) << 5;
  if (n_end > W_) n_end = W_;

  for (int n0 = 0; n0 < n_end; n0 += 32) {
    // ---- GEMM1: S^T tiles (keys n0..n0+15 and n0+16..n0+31) x 16 queries ----
    v8f s0 = {}, s1 = {};
#pragma unroll
    for (int c = 0; c < 4; ++c) {
      const int db = 32 * c + 8 * h;
      Frag k0, k1;
      const uint4* p0 = (const uint4*)(sK + (n0 + idx) * SKS + db);
      k0.q[0] = p0[0]; k0.q[1] = p0[2];
      const uint4* p1 = (const uint4*)(sK + (n0 + 16 + idx) * SKS + db);
      k1.q[0] = p1[0]; k1.q[1] = p1[2];
      s0 = __builtin_amdgcn_wmma_f32_16x16x32_bf16(false, k0.v, false, qa[c],
                                                   (short)0, s0, false, false);
      s1 = __builtin_amdgcn_wmma_f32_16x16x32_bf16(false, k1.v, false, qa[c],
                                                   (short)0, s1, false, false);
    }
    // scale + mask (mask only affects query rows < W_)
#pragma unroll
    for (int r = 0; r < 8; ++r) { s0[r] *= scale; s1[r] *= scale; }
    if (q0 < W_) {
      const int jb = n0 + 8 * h;   // key index of s0[r] is jb+r; s1: jb+16+r
#pragma unroll
      for (int r = 0; r < 8; ++r) {
        if (qrow < jb + r)      s0[r] = -1e30f;
        if (qrow < jb + 16 + r) s1[r] = -1e30f;
      }
    }
    // ---- online softmax (per query = per lane mod 16) ----
    float cm = -1e30f;
#pragma unroll
    for (int r = 0; r < 8; ++r) cm = fmaxf(cm, fmaxf(s0[r], s1[r]));
    cm = fmaxf(cm, __shfl_xor(cm, 16, 32));
    const float m_new = fmaxf(m_i, cm);
    const float alpha = __expf(m_i - m_new);
    float rs = 0.0f;
#pragma unroll
    for (int r = 0; r < 8; ++r) {
      s0[r] = __expf(s0[r] - m_new); rs += s0[r];
      s1[r] = __expf(s1[r] - m_new); rs += s1[r];
    }
    rs += __shfl_xor(rs, 16, 32);
    l_i = l_i * alpha + rs;
    m_i = m_new;

    // P^T (C layout) -> bf16 B fragment: purely per-lane packing.
    Frag pb;
#pragma unroll
    for (int e = 0; e < 8; ++e) {
      pb.s[e]     = f2bf_u(s0[e]);
      pb.s[8 + e] = f2bf_u(s1[e]);
    }

    // rescale O^T accumulators (query dim == lane dim, so scalar per lane)
#pragma unroll
    for (int t = 0; t < 8; ++t) {
#pragma unroll
      for (int r = 0; r < 8; ++r) o[t][r] *= alpha;
    }
    // ---- GEMM2: O^T[d][m] += V^T[d][keys] * P^T[keys][m] ----
#pragma unroll
    for (int t = 0; t < 8; ++t) {
      Frag vf;
      const uint4* pv = (const uint4*)(sVT + (16 * t + idx) * SVS + n0 + 8 * h);
      vf.q[0] = pv[0]; vf.q[1] = pv[2];
      o[t] = __builtin_amdgcn_wmma_f32_16x16x32_bf16(false, vf.v, false, pb.v,
                                                     (short)0, o[t], false, false);
    }
  }

  // ---- epilogue: normalize and store (lane writes 32B contiguous per tile) ----
  const float inv = 1.0f / l_i;
  float* og = O + head + (size_t)qrow * D_;
#pragma unroll
  for (int t = 0; t < 8; ++t) {
    float4 a = make_float4(o[t][0] * inv, o[t][1] * inv, o[t][2] * inv, o[t][3] * inv);
    float4 b = make_float4(o[t][4] * inv, o[t][5] * inv, o[t][6] * inv, o[t][7] * inv);
    *(float4*)(og + 16 * t + 8 * h)     = a;
    *(float4*)(og + 16 * t + 8 * h + 4) = b;
  }
}

extern "C" void kernel_launch(void* const* d_in, const int* in_sizes, int n_in,
                              void* d_out, int out_size, void* d_ws, size_t ws_size,
                              hipStream_t stream) {
  (void)in_sizes; (void)n_in; (void)out_size; (void)d_ws; (void)ws_size;
  const float* q = (const float*)d_in[0];
  const float* k = (const float*)d_in[1];
  const float* v = (const float*)d_in[2];
  float* out = (float*)d_out;
  const dim3 grid(L_ / BQ, 4 * 16);           // (32 query tiles, B*H heads)
  const size_t smem = (size_t)(W_ * SKS + D_ * SVS) * sizeof(unsigned short); // ~266 KB
  swa_wmma_kernel<<<grid, 256, smem, stream>>>(q, k, v, out);
}
